// MoEKernel_45595372814982
// MI455X (gfx1250) — matrix operand order repaired and checked by reading, not compiled, not run on hardware
//
#include <hip/hip_runtime.h>
#include <hip/hip_bf16.h>
#include <math.h>

// Problem sizes (fixed by the reference)
#define B_  8
#define T_  4096
#define D_  1024
#define H_  2048
#define E_  8
#define SP_ 64
#define S_  512   // E * SP

typedef __attribute__((ext_vector_type(16))) __bf16 v16bf;
typedef __attribute__((ext_vector_type(8)))  float  v8f;

// GEMM tiling: block = 128(M) x 128(N), K-step 32, 256 threads = 8 waves,
// wave grid 4(M) x 2(N); each wave owns 32x64 = 2x4 WMMA 16x16 tiles.
#define TM 128
#define TN 128
#define TK 32
#define LDA_PAD (TK + 8)   // conflict-free A fragment reads
#define LDB_PAD (TN + 8)   // conflict-free B fragment reads

__device__ __forceinline__ float gelu_exact(float x) {
    return 0.5f * x * (1.0f + erff(x * 0.70710678118654752440f));
}

__device__ __forceinline__ unsigned lds_off(const void* p) {
    // generic pointer = {LDS aperture, 32-bit LDS offset}; low 32 bits are the DS offset
    return (unsigned)(unsigned long long)p;
}

// async global->LDS 16B per lane, tracked with ASYNCcnt (CDNA5 async-to-LDS path)
#define ASYNC_B128(ldso, gaddr) \
    asm volatile("global_load_async_to_lds_b128 %0, %1, off" :: "v"(ldso), "v"(gaddr) : "memory")
#define WAIT_ASYNC_4() asm volatile("s_wait_asynccnt 0x4" ::: "memory")
#define WAIT_ASYNC_0() asm volatile("s_wait_asynccnt 0x0" ::: "memory")

// Generic batched GEMM: C[z] = A[z](MxK,bf16,rm) * B[z](KxN,bf16,rm)
// CSCATTER: write C rows expert-major: (E, B*SP, ldc) with z = batch b
// BGATHER : read  B rows from expert-major (E, B*SP, ldb) with z = batch b
template<bool OUT_BF16, bool GELU, bool HAS_BIAS, bool CSCATTER, bool BGATHER>
__global__ __launch_bounds__(256) void gemm_bf16_wmma(
    const __bf16* __restrict__ A, const __bf16* __restrict__ Bm,
    void* __restrict__ Cv, const float* __restrict__ bias,
    int K, int lda, int ldb, int ldc,
    long long sA, long long sB, long long sC, long long sBias)
{
    __shared__ __bf16 lA[2][TM][LDA_PAD];   // 2 x 128 x 40  (20.5 KB)
    __shared__ __bf16 lB[2][TK][LDB_PAD];   // 2 x  32 x 136 (17.4 KB)

    const int tid  = threadIdx.x;
    const int lane = tid & 31;
    const int wave = tid >> 5;
    const int wm   = wave & 3;       // 4 M-strips of 32 rows
    const int wn   = wave >> 2;      // 2 N-strips of 64 cols

    const int m0 = blockIdx.y * TM;
    const int n0 = blockIdx.x * TN;
    const int z  = blockIdx.z;

    A += (long long)z * sA;
    if (!BGATHER) Bm += (long long)z * sB;

    // tile-load mappings: 2 x 16B chunks per thread for A and for B
    const int arow = tid >> 2;           // 0..63 (+64)
    const int acol = (tid & 3) * 8;      // 0,8,16,24
    const int brow = tid >> 4;           // 0..15 (+16)
    const int bcol = (tid & 15) * 8;     // 0..120

    // per-thread global byte addresses, strided per K-step
    unsigned long long ga0 = (unsigned long long)(A + (long long)(m0 + arow)      * lda + acol);
    unsigned long long ga1 = (unsigned long long)(A + (long long)(m0 + arow + 64) * lda + acol);
    unsigned long long gb0 = (unsigned long long)(Bm + (long long)brow        * ldb + n0 + bcol);
    unsigned long long gb1 = (unsigned long long)(Bm + (long long)(brow + 16) * ldb + n0 + bcol);
    const unsigned long long aStep = (unsigned long long)TK * 2;
    const unsigned long long bStep = (unsigned long long)TK * ldb * 2;

    unsigned laOff[2], la1Off[2], lbOff[2], lb1Off[2];
    #pragma unroll
    for (int b = 0; b < 2; ++b) {
        laOff[b]  = lds_off(&lA[b][arow][acol]);
        la1Off[b] = lds_off(&lA[b][arow + 64][acol]);
        lbOff[b]  = lds_off(&lB[b][brow][bcol]);
        lb1Off[b] = lds_off(&lB[b][brow + 16][bcol]);
    }

    auto bGatherAddr = [&](int kk) -> unsigned long long {
        // B row kk = slot s -> expert-major row
        const long long gr = (long long)(kk >> 6) * (B_ * SP_) + (long long)z * SP_ + (kk & 63);
        return (unsigned long long)(Bm + gr * ldb + n0 + bcol);
    };

    auto issue = [&](int buf, int k0) {
        ASYNC_B128(laOff[buf],  ga0);
        ASYNC_B128(la1Off[buf], ga1);
        if (BGATHER) {
            ASYNC_B128(lbOff[buf],  bGatherAddr(k0 + brow));
            ASYNC_B128(lb1Off[buf], bGatherAddr(k0 + brow + 16));
        } else {
            ASYNC_B128(lbOff[buf],  gb0);
            ASYNC_B128(lb1Off[buf], gb1);
            gb0 += bStep; gb1 += bStep;
        }
        ga0 += aStep; ga1 += aStep;
    };

    const int mlane = lane & 15;
    const int half  = lane >> 4;

    v8f acc[2][4];
    #pragma unroll
    for (int i = 0; i < 2; ++i)
        #pragma unroll
        for (int j = 0; j < 4; ++j)
            #pragma unroll
            for (int v = 0; v < 8; ++v) acc[i][j][v] = 0.0f;

    auto compute = [&](int buf) {
        union Frag { v16bf v; unsigned u[8]; } af[2], bfr[4];
        #pragma unroll
        for (int mi = 0; mi < 2; ++mi) {
            const __bf16* ap = &lA[buf][wm * 32 + mi * 16 + mlane][0];
            #pragma unroll
            for (int v = 0; v < 8; ++v) {
                // A 16x32 bf16: lanes 0-15 hold K {0-7,16-23}, lanes 16-31 hold K {8-15,24-31}
                const int kk = (v >> 2) * 16 + half * 8 + (v & 3) * 2;
                af[mi].u[v] = *(const unsigned*)(ap + kk);
            }
        }
        #pragma unroll
        for (int ni = 0; ni < 4; ++ni) {
            const __bf16* bp = &lB[buf][0][wn * 64 + ni * 16 + mlane];  // B 32x16: lane l = column l%16, K in same order as A
            #pragma unroll
            for (int v = 0; v < 8; ++v) {
                const int kk = (v >> 2) * 16 + half * 8 + (v & 3) * 2;
                bfr[ni].v[2 * v]     = bp[kk * LDB_PAD];
                bfr[ni].v[2 * v + 1] = bp[(kk + 1) * LDB_PAD];
            }
        }
        #pragma unroll
        for (int mi = 0; mi < 2; ++mi)
            #pragma unroll
            for (int ni = 0; ni < 4; ++ni)
                acc[mi][ni] = __builtin_amdgcn_wmma_f32_16x16x32_bf16(
                    false, af[mi].v, false, bfr[ni].v,
                    (short)0, acc[mi][ni], false, false);
    };

    // Software pipeline: nk is even at every call site; buffer index is
    // compile-time constant in each half of the pair-loop (no selects).
    const int nk = K / TK;
    issue(0, 0);
    issue(1, TK);

    #pragma unroll 1
    for (int i = 0; i < nk; i += 2) {
        const bool more = (i + 2) < nk;

        // tile i in buffer 0 (tile i+1 still outstanding -> wait to 4)
        WAIT_ASYNC_4();
        __syncthreads();
        compute(0);
        __syncthreads();
        if (more) issue(0, (i + 2) * TK);

        // tile i+1 in buffer 1
        if (more) { WAIT_ASYNC_4(); } else { WAIT_ASYNC_0(); }
        __syncthreads();
        compute(1);
        __syncthreads();
        if (more) issue(1, (i + 3) * TK);
    }

    // Epilogue. C tile layout: lane -> n = lane%16; VGPR v -> m = v + 8*(lane/16)
    #pragma unroll
    for (int mi = 0; mi < 2; ++mi) {
        #pragma unroll
        for (int ni = 0; ni < 4; ++ni) {
            const int c = n0 + wn * 64 + ni * 16 + mlane;
            float bv = 0.0f;
            if (HAS_BIAS) bv = bias[(long long)z * sBias + c];
            #pragma unroll
            for (int v = 0; v < 8; ++v) {
                const int r = m0 + wm * 32 + mi * 16 + v + 8 * half;
                float val = acc[mi][ni][v] + bv;
                if (GELU) val = gelu_exact(val);
                long long off;
                if (CSCATTER)
                    off = ((long long)(r >> 6) * (B_ * SP_) + (long long)z * SP_ + (r & 63))
                          * ldc + c;
                else
                    off = (long long)z * sC + (long long)r * ldc + c;
                if (OUT_BF16) ((__bf16*)Cv)[off] = (__bf16)val;
                else          ((float*)Cv)[off]  = val;
            }
        }
    }
}

// ---- elementwise / softmax helpers ----

__global__ void cvt_bf16(const float* __restrict__ src, __bf16* __restrict__ dst, long long n) {
    long long i = (long long)blockIdx.x * 256 + threadIdx.x;
    if (i < n) dst[i] = (__bf16)src[i];
}

// WrT[k*S + s] = Wr[s*D + k]
__global__ void transpose_wr(const float* __restrict__ Wr, __bf16* __restrict__ WrT) {
    const int idx = blockIdx.x * 256 + threadIdx.x;   // D_*S_ total
    const int k = idx / S_;
    const int s = idx - k * S_;
    WrT[idx] = (__bf16)Wr[(long long)s * D_ + k];
}

__global__ void expand_bias(const float* __restrict__ eb, float* __restrict__ sbias) {
    const int s = blockIdx.x * 256 + threadIdx.x;
    if (s < S_) sbias[s] = eb[s >> 6];
}

// dispatch = softmax over tokens (axis T); write transposed bf16 (B,S,T)
__global__ __launch_bounds__(256) void softmax_over_T(
    const float* __restrict__ logits, __bf16* __restrict__ dispT)
{
    __shared__ float red[256];
    const int b = blockIdx.x / S_;
    const int s = blockIdx.x - b * S_;
    const int tid = threadIdx.x;
    const float* col = logits + (long long)b * T_ * S_ + s;

    float v[16];
    float mx = -3.4e38f;
    #pragma unroll
    for (int i = 0; i < 16; ++i) {
        v[i] = col[(long long)(tid + 256 * i) * S_];
        mx = fmaxf(mx, v[i]);
    }
    red[tid] = mx; __syncthreads();
    for (int off = 128; off > 0; off >>= 1) {
        if (tid < off) red[tid] = fmaxf(red[tid], red[tid + off]);
        __syncthreads();
    }
    mx = red[0]; __syncthreads();

    float sum = 0.0f;
    #pragma unroll
    for (int i = 0; i < 16; ++i) { v[i] = __expf(v[i] - mx); sum += v[i]; }
    red[tid] = sum; __syncthreads();
    for (int off = 128; off > 0; off >>= 1) {
        if (tid < off) red[tid] += red[tid + off];
        __syncthreads();
    }
    const float inv = 1.0f / red[0];

    __bf16* dst = dispT + ((long long)b * S_ + s) * T_;
    #pragma unroll
    for (int i = 0; i < 16; ++i) dst[tid + 256 * i] = (__bf16)(v[i] * inv);
}

// combine = softmax over slots (axis S); bf16 (B,T,S)
__global__ __launch_bounds__(256) void softmax_over_S(
    const float* __restrict__ logits, __bf16* __restrict__ comb)
{
    __shared__ float red[256];
    const long long row = blockIdx.x;          // b*T + t
    const int tid = threadIdx.x;
    const float* p = logits + row * S_;

    float v0 = p[tid], v1 = p[tid + 256];
    red[tid] = fmaxf(v0, v1); __syncthreads();
    for (int off = 128; off > 0; off >>= 1) {
        if (tid < off) red[tid] = fmaxf(red[tid], red[tid + off]);
        __syncthreads();
    }
    const float mx = red[0]; __syncthreads();

    const float e0 = __expf(v0 - mx), e1 = __expf(v1 - mx);
    red[tid] = e0 + e1; __syncthreads();
    for (int off = 128; off > 0; off >>= 1) {
        if (tid < off) red[tid] += red[tid + off];
        __syncthreads();
    }
    const float inv = 1.0f / red[0];

    __bf16* dst = comb + row * S_;
    dst[tid]       = (__bf16)(e0 * inv);
    dst[tid + 256] = (__bf16)(e1 * inv);
}

// ---- host orchestration ----

extern "C" void kernel_launch(void* const* d_in, const int* in_sizes, int n_in,
                              void* d_out, int out_size, void* d_ws, size_t ws_size,
                              hipStream_t stream) {
    const float* x  = (const float*)d_in[0];
    const float* Wr = (const float*)d_in[1];
    const float* eb = (const float*)d_in[2];
    const float* W1 = (const float*)d_in[3];
    const float* b1 = (const float*)d_in[4];
    const float* W2 = (const float*)d_in[5];
    const float* b2 = (const float*)d_in[6];
    float* out = (float*)d_out;

    char* p = (char*)d_ws;
    auto carve = [&](size_t bytes) -> void* {
        void* r = (void*)p;
        p += (bytes + 255) & ~(size_t)255;
        return r;
    };

    __bf16* xb       = (__bf16*)carve((size_t)B_ * T_ * D_ * 2);   //  67 MB
    __bf16* WrT      = (__bf16*)carve((size_t)D_ * S_ * 2);        //   1 MB
    __bf16* W1b      = (__bf16*)carve((size_t)E_ * D_ * H_ * 2);   //  34 MB
    __bf16* W2b      = (__bf16*)carve((size_t)E_ * H_ * D_ * 2);   //  34 MB
    float*  logits   = (float*) carve((size_t)B_ * T_ * S_ * 4);   //  67 MB
    float*  sbias    = (float*) carve((size_t)S_ * 4);
    __bf16* dispT    = (__bf16*)carve((size_t)B_ * S_ * T_ * 2);   //  34 MB
    __bf16* comb     = (__bf16*)carve((size_t)B_ * T_ * S_ * 2);   //  34 MB
    __bf16* slotinE  = (__bf16*)carve((size_t)E_ * B_ * SP_ * D_ * 2);  // 8 MB (E, B*SP, D)
    __bf16* hbufE    = (__bf16*)carve((size_t)E_ * B_ * SP_ * H_ * 2);  // 17 MB (E, B*SP, H)
    __bf16* slotoutE = (__bf16*)carve((size_t)E_ * B_ * SP_ * D_ * 2);  // 8 MB (E, B*SP, D)

    // 1) precision converts
    { long long n = (long long)B_ * T_ * D_;
      cvt_bf16<<<(unsigned)((n + 255) / 256), 256, 0, stream>>>(x, xb, n); }
    transpose_wr<<<(D_ * S_) / 256, 256, 0, stream>>>(Wr, WrT);
    { long long n = (long long)E_ * D_ * H_;
      cvt_bf16<<<(unsigned)((n + 255) / 256), 256, 0, stream>>>(W1, W1b, n); }
    { long long n = (long long)E_ * H_ * D_;
      cvt_bf16<<<(unsigned)((n + 255) / 256), 256, 0, stream>>>(W2, W2b, n); }
    expand_bias<<<2, 256, 0, stream>>>(eb, sbias);

    // 2) router logits (fp32, + per-slot bias; TAU == 1):  (BT x S) = xb * WrT
    gemm_bf16_wmma<false, false, true, false, false>
        <<<dim3(S_ / TN, (B_ * T_) / TM, 1), 256, 0, stream>>>(
        xb, WrT, logits, sbias, D_, D_, S_, S_, 0, 0, 0, 0);

    // 3) softmaxes
    softmax_over_T<<<B_ * S_, 256, 0, stream>>>(logits, dispT);
    softmax_over_S<<<B_ * T_, 256, 0, stream>>>(logits, comb);

    // 4) slot_in[b] = dispatchT[b](SxT) * x[b](TxD), scattered to expert-major (E,B*SP,D)
    gemm_bf16_wmma<true, false, false, true, false>
        <<<dim3(D_ / TN, S_ / TM, B_), 256, 0, stream>>>(
        dispT, xb, slotinE, nullptr, T_, T_, D_, D_,
        (long long)S_ * T_, (long long)T_ * D_, 0, 0);

    // 5) grouped FFN, single launch per layer, batched over experts (M = B*SP = 512)
    gemm_bf16_wmma<true, true, true, false, false>
        <<<dim3(H_ / TN, (B_ * SP_) / TM, E_), 256, 0, stream>>>(
        slotinE, W1b, hbufE, b1, D_, D_, H_, H_,
        (long long)B_ * SP_ * D_, (long long)D_ * H_, (long long)B_ * SP_ * H_, H_);

    gemm_bf16_wmma<true, false, true, false, false>
        <<<dim3(D_ / TN, (B_ * SP_) / TM, E_), 256, 0, stream>>>(
        hbufE, W2b, slotoutE, b2, H_, H_, D_, D_,
        (long long)B_ * SP_ * H_, (long long)H_ * D_, (long long)B_ * SP_ * D_, D_);

    // 6) out[b] = combine[b](TxS) * slot_out[b](SxD), gathering B rows from expert-major
    gemm_bf16_wmma<false, false, false, false, true>
        <<<dim3(D_ / TN, T_ / TM, B_), 256, 0, stream>>>(
        comb, slotoutE, out, nullptr, S_, S_, D_, D_,
        (long long)T_ * S_, 0, (long long)T_ * D_, 0);
}
